// RPN_24043226923933
// MI455X (gfx1250) — matrix operand order repaired and checked by reading, not compiled, not run on hardware
//
#include <hip/hip_runtime.h>
#include <hip/hip_bf16.h>
#include <math.h>

typedef __attribute__((ext_vector_type(16))) _Float16 v16h;
typedef __attribute__((ext_vector_type(8)))  _Float16 v8h;
typedef __attribute__((ext_vector_type(8)))  float    v8f;

#define FH 64
#define FW 64
#define CI 1024
#define CO 512
#define NANCH 36864      // 64*64*9
#define NPAD  65536
#define NPRE  12000
#define NPOST 2000
#define NWORD 375        // 12000/32

// ---------------------------------------------------------------------------
// Packing kernels: convert f32 inputs to WMMA-friendly f16 layouts.
// ---------------------------------------------------------------------------
__global__ void pack_x_kernel(const float* __restrict__ x, _Float16* __restrict__ xT) {
    int i = blockIdx.x * blockDim.x + threadIdx.x;           // over 1024*64*64
    if (i >= CI * FH * FW) return;
    int ci  = i / (FH * FW);
    int sp  = i % (FH * FW);                                  // y*64+x
    xT[(size_t)sp * CI + ci] = (_Float16)x[i];
}

__global__ void pack_w_kernel(const float* __restrict__ w, _Float16* __restrict__ wT) {
    int i = blockIdx.x * blockDim.x + threadIdx.x;           // over 512*1024*9
    if (i >= CO * CI * 9) return;
    int co  = i / (CI * 9);
    int r   = i % (CI * 9);
    int ci  = r / 9;
    int tap = r % 9;
    wT[(size_t)tap * CO * CI + (size_t)co * CI + ci] = (_Float16)w[i];
}

__global__ void pack_wls_kernel(const float* __restrict__ lw, const float* __restrict__ sw,
                                _Float16* __restrict__ wLS) {
    int i = blockIdx.x * blockDim.x + threadIdx.x;           // over 64*512
    if (i >= 64 * 512) return;
    int n = i / 512, k = i % 512;
    float v = 0.f;
    if (n < 36)      v = lw[n * 512 + k];
    else if (n < 54) v = sw[(n - 36) * 512 + k];
    wLS[i] = (_Float16)v;
}

// ---------------------------------------------------------------------------
// 3x3 conv as implicit GEMM on WMMA.  Block: one image row y (BM=64 spatial),
// 128 output channels (BN=128).  8 waves; wave (wm, wn) computes a 16x64 tile.
// A tile is double-buffered in LDS and filled by the CDNA5 async-DMA path
// (global_load_async_to_lds_b128, ASYNCcnt), overlapping the next chunk's
// transfer with the current chunk's WMMAs.
// ---------------------------------------------------------------------------
__global__ __launch_bounds__(256) void conv3x3_wmma_kernel(
        const _Float16* __restrict__ xT,   // [4096][1024]
        const _Float16* __restrict__ wT,   // [9][512][1024]
        const float*    __restrict__ cb,   // [512]
        _Float16*       __restrict__ hbuf) // [4096][512]
{
    __shared__ _Float16 Asm[2][64][40];    // pitch 40 halfs = 80B (16B aligned rows)
    const int y      = blockIdx.y;
    const int coBase = blockIdx.x * 128;
    const int t      = threadIdx.x;
    const int wave   = t >> 5;
    const int lane   = t & 31;
    const int wm     = wave & 3;           // M sub-tile (16 rows)
    const int wn     = wave >> 2;          // 0..1 -> 64 columns each
    const int lrow   = lane & 15;
    const int hi     = lane >> 4;          // 0 | 1

    v8f acc[4];
    #pragma unroll
    for (int f = 0; f < 4; ++f)
        #pragma unroll
        for (int e = 0; e < 8; ++e) acc[f][e] = 0.f;

    const int mStage  = t >> 2;            // 0..63  (spatial x within row)
    const int kkStage = (t & 3) * 8;       // 0,8,16,24

    unsigned ldsOff0 = (unsigned)(size_t)&Asm[0][mStage][kkStage];
    unsigned ldsOff1 = (unsigned)(size_t)&Asm[1][mStage][kkStage];

    for (int tap = 0; tap < 9; ++tap) {
        const int dy = tap / 3 - 1, dx = tap % 3 - 1;
        const int ysrc = y + dy;
        if (ysrc < 0 || ysrc >= FH) continue;          // uniform: all-zero A tile
        const int xsrc = mStage + dx;
        const bool ok = (xsrc >= 0) && (xsrc < FW);
        const _Float16* src = xT + (size_t)(ysrc * FW + xsrc) * CI + kkStage;

        // Prologue: stage chunk 0 into buffer 0 (buffer 0 reads from the
        // previous tap retired at that tap's c==31 top barrier).
        if (ok) {
            unsigned long long ga = (unsigned long long)(size_t)src;
            asm volatile("global_load_async_to_lds_b128 %0, %1, off\n\t"
                         "s_wait_xcnt 0x0"
                         :: "v"(ldsOff0), "v"(ga) : "memory");
        } else {
            v8h z;
            #pragma unroll
            for (int e = 0; e < 8; ++e) z[e] = (_Float16)0.f;
            *(v8h*)&Asm[0][mStage][kkStage] = z;
        }

        for (int c = 0; c < 32; ++c) {
            const int cur = c & 1, nxt = cur ^ 1;
            __syncthreads();   // all waves finished reading buf[nxt]'s old data
            if (c < 31) {
                if (ok) {
                    unsigned long long ga =
                        (unsigned long long)(size_t)(src + (size_t)(c + 1) * 32);
                    unsigned dst = nxt ? ldsOff1 : ldsOff0;
                    asm volatile("global_load_async_to_lds_b128 %0, %1, off\n\t"
                                 "s_wait_xcnt 0x0"
                                 :: "v"(dst), "v"(ga) : "memory");
                } else {
                    v8h z;
                    #pragma unroll
                    for (int e = 0; e < 8; ++e) z[e] = (_Float16)0.f;
                    *(v8h*)&Asm[nxt][mStage][kkStage] = z;
                }
                asm volatile("s_wait_asynccnt 0x1" ::: "memory"); // chunk c landed
            } else {
                asm volatile("s_wait_asynccnt 0x0" ::: "memory");
            }
            __syncthreads();   // chunk c visible to every wave

            // A fragment: lane row = wm*16 + lrow; K runs {hi*16..+15}
            const _Float16* ar = &Asm[cur][wm * 16 + lrow][hi * 16];
            v8h alo = *(const v8h*)ar;
            v8h ahi = *(const v8h*)(ar + 8);
            v16h a;
            #pragma unroll
            for (int e = 0; e < 8; ++e) { a[e] = alo[e]; a[e + 8] = ahi[e]; }

            const int ci0 = c * 32;
            const _Float16* wbase = wT + (size_t)tap * CO * CI + ci0 + hi * 16;
            #pragma unroll
            for (int nf = 0; nf < 4; ++nf) {
                const int co = coBase + wn * 64 + nf * 16 + lrow;
                v16h b = *(const v16h*)(wbase + (size_t)co * CI);
                acc[nf] = __builtin_amdgcn_wmma_f32_16x16x32_f16(
                    false, a, false, b, (short)0, acc[nf], false, false);
            }
        }
    }

    // Epilogue: bias + leaky relu, store f16 activation [m][co]
    #pragma unroll
    for (int nf = 0; nf < 4; ++nf) {
        const int co = coBase + wn * 64 + nf * 16 + lrow;
        const float bsc = cb[co];
        #pragma unroll
        for (int r = 0; r < 8; ++r) {
            const int mx = wm * 16 + r + hi * 8;
            float v = acc[nf][r] + bsc;
            v = v > 0.f ? v : 0.01f * v;
            hbuf[(size_t)(y * FW + mx) * CO + co] = (_Float16)v;
        }
    }
}

// ---------------------------------------------------------------------------
// 1x1 head convs (loc 36ch + score 18ch packed into N=64) as one WMMA GEMM.
// ---------------------------------------------------------------------------
__global__ __launch_bounds__(256) void heads_wmma_kernel(
        const _Float16* __restrict__ hbuf, // [4096][512]
        const _Float16* __restrict__ wLS,  // [64][512]
        const float* __restrict__ lb, const float* __restrict__ sb,
        float* __restrict__ outL, float* __restrict__ outS)
{
    const int m0   = blockIdx.x * 64;
    const int t    = threadIdx.x;
    const int wave = t >> 5, lane = t & 31;
    const int wm   = wave >> 1;
    const int wn   = (wave & 1) * 32;
    const int lrow = lane & 15, hi = lane >> 4;

    v8f acc[2];
    #pragma unroll
    for (int f = 0; f < 2; ++f)
        #pragma unroll
        for (int e = 0; e < 8; ++e) acc[f][e] = 0.f;

    const _Float16* arow = hbuf + (size_t)(m0 + wm * 16 + lrow) * CO + hi * 16;
    for (int k0 = 0; k0 < CO; k0 += 32) {
        v8h alo = *(const v8h*)(arow + k0);
        v8h ahi = *(const v8h*)(arow + k0 + 8);
        v16h a;
        #pragma unroll
        for (int e = 0; e < 8; ++e) { a[e] = alo[e]; a[e + 8] = ahi[e]; }
        #pragma unroll
        for (int nf = 0; nf < 2; ++nf) {
            const int n = wn + nf * 16 + lrow;
            v16h b = *(const v16h*)(wLS + (size_t)n * CO + k0 + hi * 16);
            acc[nf] = __builtin_amdgcn_wmma_f32_16x16x32_f16(
                false, a, false, b, (short)0, acc[nf], false, false);
        }
    }

    #pragma unroll
    for (int nf = 0; nf < 2; ++nf) {
        const int n = wn + nf * 16 + lrow;
        #pragma unroll
        for (int r = 0; r < 8; ++r) {
            const int m = m0 + wm * 16 + r + hi * 8;
            float v = acc[nf][r];
            if (n < 36)      outL[(size_t)m * 36 + n] = v + lb[n];
            else if (n < 54) outS[(size_t)m * 18 + (n - 36)] = v + sb[n - 36];
        }
    }
}

// ---------------------------------------------------------------------------
// Per-anchor decode: anchors, softmax fg, box de-parameterize, clip, sort key.
// ---------------------------------------------------------------------------
__global__ void decode_kernel(const float* __restrict__ locs, const float* __restrict__ scores,
                              float* __restrict__ anchorsOut, float* __restrict__ roisAll,
                              unsigned long long* __restrict__ keys,
                              const int* __restrict__ ih, const int* __restrict__ iw)
{
    int i = blockIdx.x * blockDim.x + threadIdx.x;
    if (i >= NPAD) return;
    if (i >= NANCH) { keys[i] = ~0ull; return; }

    const int m = i / 9, a = i % 9;
    const int y = m >> 6, x = m & 63;
    const int ridx = a / 3, sidx = a % 3;
    const float rr = (ridx == 0) ? 0.5f : ((ridx == 1) ? 1.0f : 2.0f);
    const float ss = (sidx == 0) ? 8.f : ((sidx == 1) ? 16.f : 32.f);
    const float hh = 16.f * ss * sqrtf(rr);
    const float ww = 16.f * ss * sqrtf(1.f / rr);
    const float py = y * 16.f + 8.f, px = x * 16.f + 8.f;
    const float a0 = py - 0.5f * hh, a1 = px - 0.5f * ww;
    const float a2 = py + 0.5f * hh, a3 = px + 0.5f * ww;
    anchorsOut[i * 4 + 0] = a0; anchorsOut[i * 4 + 1] = a1;
    anchorsOut[i * 4 + 2] = a2; anchorsOut[i * 4 + 3] = a3;

    const float s0 = scores[(size_t)m * 18 + a * 2];
    const float s1 = scores[(size_t)m * 18 + a * 2 + 1];
    const float fg = 1.f / (1.f + expf(s0 - s1));

    const float l0 = locs[(size_t)m * 36 + a * 4 + 0] * 0.1f;
    const float l1 = locs[(size_t)m * 36 + a * 4 + 1] * 0.1f;
    const float l2 = locs[(size_t)m * 36 + a * 4 + 2] * 0.2f;
    const float l3 = locs[(size_t)m * 36 + a * 4 + 3] * 0.2f;

    const float ha = a2 - a0, wa = a3 - a1;
    const float cya = a0 + 0.5f * ha, cxa = a1 + 0.5f * wa;
    const float cy = l0 * ha + cya, cx = l1 * wa + cxa;
    const float hb = expf(l2) * ha, wb = expf(l3) * wa;
    const float H = (float)ih[0], W = (float)iw[0];
    const float r0 = fminf(fmaxf(cy - 0.5f * hb, 0.f), H);
    const float r1 = fminf(fmaxf(cx - 0.5f * wb, 0.f), W);
    const float r2 = fminf(fmaxf(cy + 0.5f * hb, 0.f), H);
    const float r3 = fminf(fmaxf(cx + 0.5f * wb, 0.f), W);
    roisAll[i * 4 + 0] = r0; roisAll[i * 4 + 1] = r1;
    roisAll[i * 4 + 2] = r2; roisAll[i * 4 + 3] = r3;

    // fg in (0,1): positive, so IEEE bit pattern is order-preserving.
    unsigned fb = __float_as_uint(fg);
    keys[i] = (((unsigned long long)(~fb)) << 32) | (unsigned)i;   // ascending => best first, stable
}

// ---------------------------------------------------------------------------
// Bitonic sort stage (ascending over 65536 u64 keys).
// ---------------------------------------------------------------------------
__global__ void bitonic_kernel(unsigned long long* __restrict__ keys, int j, int k) {
    int i = blockIdx.x * blockDim.x + threadIdx.x;
    int l = i ^ j;
    if (l > i) {
        unsigned long long a = keys[i], b = keys[l];
        const bool up = ((i & k) == 0);
        if ((a > b) == up) { keys[i] = b; keys[l] = a; }
    }
}

__global__ void gather_top_kernel(const unsigned long long* __restrict__ keys,
                                  const float* __restrict__ roisAll,
                                  float* __restrict__ roisTop, unsigned* __restrict__ valid)
{
    int p = blockIdx.x * blockDim.x + threadIdx.x;
    if (p >= NPRE) return;
    const unsigned idx = (unsigned)(keys[p] & 0xffffffffull);
    const float r0 = roisAll[(size_t)idx * 4 + 0];
    const float r1 = roisAll[(size_t)idx * 4 + 1];
    const float r2 = roisAll[(size_t)idx * 4 + 2];
    const float r3 = roisAll[(size_t)idx * 4 + 3];
    roisTop[p * 4 + 0] = r0; roisTop[p * 4 + 1] = r1;
    roisTop[p * 4 + 2] = r2; roisTop[p * 4 + 3] = r3;
    valid[p] = (((r2 - r0) >= 16.f) && ((r3 - r1) >= 16.f)) ? 1u : 0u;
}

// IoU suppression bitmask: thread -> (row p, word b); 32 IoUs per thread.
__global__ void iou_mask_kernel(const float* __restrict__ roisTop, unsigned* __restrict__ maskM) {
    int tid = blockIdx.x * blockDim.x + threadIdx.x;
    if (tid >= NPRE * NWORD) return;
    const int p = tid / NWORD, b = tid % NWORD;
    const float py1 = roisTop[p * 4 + 0], px1 = roisTop[p * 4 + 1];
    const float py2 = roisTop[p * 4 + 2], px2 = roisTop[p * 4 + 3];
    const float pa = (py2 - py1) * (px2 - px1);
    unsigned w = 0;
    const int qb = b * 32;
    #pragma unroll 4
    for (int j = 0; j < 32; ++j) {
        const int q = qb + j;
        if (q <= p) continue;
        const float qy1 = roisTop[q * 4 + 0], qx1 = roisTop[q * 4 + 1];
        const float qy2 = roisTop[q * 4 + 2], qx2 = roisTop[q * 4 + 3];
        const float qa = (qy2 - qy1) * (qx2 - qx1);
        const float iy = fmaxf(0.f, fminf(py2, qy2) - fmaxf(py1, qy1));
        const float ix = fmaxf(0.f, fminf(px2, qx2) - fmaxf(px1, qx1));
        const float inter = iy * ix;
        const float iou = inter / (pa + qa - inter + 1e-10f);
        if (iou > 0.7f) w |= (1u << j);
    }
    maskM[(size_t)p * NWORD + b] = w;
}

// Sequential greedy scan over the mask matrix: single wave.
__global__ void nms_scan_kernel(const unsigned* __restrict__ maskM,
                                const unsigned* __restrict__ valid, unsigned* __restrict__ keep)
{
    __shared__ unsigned supp[NWORD];
    const int lane = threadIdx.x;
    for (int b = lane; b < NWORD; b += 32) {
        unsigned w = 0;
        for (int j = 0; j < 32; ++j)
            if (!valid[b * 32 + j]) w |= (1u << j);
        supp[b] = w;
    }
    __syncthreads();
    for (int i = 0; i < NPRE; ++i) {
        const unsigned sw = supp[i >> 5];
        const bool alive = ((sw >> (i & 31)) & 1u) == 0u;
        if (lane == 0) keep[i] = alive ? 1u : 0u;
        if (alive) {
            for (int b = lane; b < NWORD; b += 32)
                supp[b] |= maskM[(size_t)i * NWORD + b];
        }
        __syncthreads();
    }
}

// Compact kept rois (score order), zero-fill tail — matches rois[sel]*ok.
__global__ void finalize_kernel(const unsigned* __restrict__ keep,
                                const float* __restrict__ roisTop, float* __restrict__ outR)
{
    const int t = threadIdx.x;
    for (int j = t; j < NPOST * 4; j += 256) outR[j] = 0.f;
    __syncthreads();
    if (t == 0) {
        int cnt = 0;
        for (int i = 0; i < NPRE && cnt < NPOST; ++i) {
            if (keep[i]) {
                outR[cnt * 4 + 0] = roisTop[i * 4 + 0];
                outR[cnt * 4 + 1] = roisTop[i * 4 + 1];
                outR[cnt * 4 + 2] = roisTop[i * 4 + 2];
                outR[cnt * 4 + 3] = roisTop[i * 4 + 3];
                ++cnt;
            }
        }
    }
}

// ---------------------------------------------------------------------------
extern "C" void kernel_launch(void* const* d_in, const int* in_sizes, int n_in,
                              void* d_out, int out_size, void* d_ws, size_t ws_size,
                              hipStream_t stream) {
    const float* x   = (const float*)d_in[0];
    const float* cw  = (const float*)d_in[1];
    const float* cb  = (const float*)d_in[2];
    const float* sw  = (const float*)d_in[3];
    const float* sb  = (const float*)d_in[4];
    const float* lw  = (const float*)d_in[5];
    const float* lb  = (const float*)d_in[6];
    const int*   ih  = (const int*)d_in[7];
    const int*   iw  = (const int*)d_in[8];

    float* outL = (float*)d_out;                    // 36864*4
    float* outS = outL + 147456;                    // 36864*2
    float* outR = outL + 221184;                    // 2000*4
    float* outA = outL + 229184;                    // 36864*4

    char* ws = (char*)d_ws;
    _Float16* xT   = (_Float16*)(ws);                               // 8,388,608 B
    _Float16* wT   = (_Float16*)(ws + 8388608);                     // 9,437,184 B
    _Float16* hbuf = (_Float16*)(ws + 17825792);                    // 4,194,304 B
    _Float16* wLS  = (_Float16*)(ws + 22020096);                    //    65,536 B
    float*    roisAll = (float*)(ws + 22085632);                    //   589,824 B
    unsigned long long* keys = (unsigned long long*)(ws + 22675456);//   524,288 B
    float*    roisTop = (float*)(ws + 23199744);                    //   192,000 B
    unsigned* valid   = (unsigned*)(ws + 23391744);                 //    48,000 B
    unsigned* keep    = (unsigned*)(ws + 23439744);                 //    48,000 B
    unsigned* maskM   = (unsigned*)(ws + 23487744);                 // 18,000,000 B

    pack_x_kernel  <<<(CI * FH * FW + 255) / 256, 256, 0, stream>>>(x, xT);
    pack_w_kernel  <<<(CO * CI * 9 + 255) / 256, 256, 0, stream>>>(cw, wT);
    pack_wls_kernel<<<(64 * 512 + 255) / 256, 256, 0, stream>>>(lw, sw, wLS);

    conv3x3_wmma_kernel<<<dim3(4, 64), 256, 0, stream>>>(xT, wT, cb, hbuf);
    heads_wmma_kernel  <<<64, 256, 0, stream>>>(hbuf, wLS, lb, sb, outL, outS);

    decode_kernel<<<NPAD / 256, 256, 0, stream>>>(outL, outS, outA, roisAll, keys, ih, iw);

    for (int k = 2; k <= NPAD; k <<= 1)
        for (int j = k >> 1; j > 0; j >>= 1)
            bitonic_kernel<<<NPAD / 256, 256, 0, stream>>>(keys, j, k);

    gather_top_kernel<<<(NPRE + 255) / 256, 256, 0, stream>>>(keys, roisAll, roisTop, valid);
    iou_mask_kernel<<<(NPRE * NWORD + 255) / 256, 256, 0, stream>>>(roisTop, maskM);
    nms_scan_kernel<<<1, 32, 0, stream>>>(maskM, valid, keep);
    finalize_kernel<<<1, 256, 0, stream>>>(keep, roisTop, outR);
}
